// SelfAttention_10101763080197
// MI455X (gfx1250) — matrix operand order repaired and checked by reading, not compiled, not run on hardware
//
#include <hip/hip_runtime.h>

// ---- MI455X (gfx1250) flash-attention, fp32 in/out, f16 WMMA compute ----
// Software-pipelined KV tiles: tile i+1's global loads overlap tile i's WMMAs.

typedef __attribute__((ext_vector_type(16))) _Float16 v16h;
typedef __attribute__((ext_vector_type(8)))  float    v8f;

#define SEQ 2048
#define NB  4
#define NH  16
#define HD  64
#define BM  128          // query rows per workgroup (8 waves x 16)
#define BN  64           // kv columns per LDS tile
#define KSTRIDE 72       // halfs per LDS row (64 + 8 pad -> conflict-free b128)
#define PSTRIDE 72
#define TSTRIDE (3 * NH * HD)   // floats between consecutive t in qkv

static __device__ __forceinline__ v8f wmma_f16(v16h a, v16h b, v8f c) {
  // v_wmma_f32_16x16x32_f16: (neg_a, A, neg_b, B, c_mod, C, reuse_a, reuse_b)
  return __builtin_amdgcn_wmma_f32_16x16x32_f16(false, a, false, b, (short)0, c,
                                                false, false);
}

// Build a v16h fragment from two 16-byte LDS chunks (VGPR j <- halfs 2j,2j+1).
static __device__ __forceinline__ v16h ld16h2(const _Float16* p0,
                                              const _Float16* p1) {
  union { v16h v; uint4 q[2]; } u;
  u.q[0] = *reinterpret_cast<const uint4*>(p0);
  u.q[1] = *reinterpret_cast<const uint4*>(p1);
  return u.v;
}

union H8 { _Float16 h[8]; uint4 u; };

__launch_bounds__(256)
__global__ void fa_fwd_kernel(const float* __restrict__ qkv,
                              const int*   __restrict__ kpm,
                              float*       __restrict__ out) {
  __shared__ __align__(16) _Float16 sK[BN * KSTRIDE];      // [s][d]   f16
  __shared__ __align__(16) _Float16 sV[HD * KSTRIDE];      // [d][s]   f16 (transposed)
  __shared__ __align__(16) _Float16 sP[8 * 16 * PSTRIDE];  // per-wave 16 x BN probs
  __shared__ float sMask[BN];

  const int b      = blockIdx.z;
  const int h      = blockIdx.y;
  const int rowblk = blockIdx.x;
  const int tid    = threadIdx.x;
  const int wave   = tid >> 5;
  const int lane   = tid & 31;
  const int hl     = lane >> 4;      // 0: lanes 0-15, 1: lanes 16-31
  const int l16    = lane & 15;

  const float LOG2E  = 1.44269504088896f;
  const float qscale = 0.125f * LOG2E;           // 1/sqrt(64) * log2(e)
  const float MADD   = -10000.0f * LOG2E;        // mask add, log2 domain

  const size_t baseBT = (size_t)b * SEQ * TSTRIDE;
  const int    tbase  = rowblk * BM + wave * 16; // this wave's 16-row strip

  // ---- Q A-fragments (layout: lane M=l16; halfs = K chunks per ISA table) ----
  v16h qa[2];
  {
    const float* qrow = qkv + baseBT + (size_t)(tbase + l16) * TSTRIDE + h * HD;
#pragma unroll
    for (int dk = 0; dk < 2; ++dk) {
      const int cb = dk * 32 + hl * 8;
      float4 c0 = reinterpret_cast<const float4*>(qrow + cb)[0];
      float4 c1 = reinterpret_cast<const float4*>(qrow + cb)[1];
      float4 c2 = reinterpret_cast<const float4*>(qrow + cb + 16)[0];
      float4 c3 = reinterpret_cast<const float4*>(qrow + cb + 16)[1];
      float f[16] = {c0.x, c0.y, c0.z, c0.w, c1.x, c1.y, c1.z, c1.w,
                     c2.x, c2.y, c2.z, c2.w, c3.x, c3.y, c3.z, c3.w};
      union { v16h v; _Float16 h[16]; } u;
#pragma unroll
      for (int e = 0; e < 16; ++e) u.h[e] = (_Float16)(f[e] * qscale);
      qa[dk] = u.v;
    }
  }

  v8f Oacc[4];
#pragma unroll
  for (int dc = 0; dc < 4; ++dc) Oacc[dc] = (v8f){0, 0, 0, 0, 0, 0, 0, 0};
  float mrun[8], lrun[8];
#pragma unroll
  for (int r = 0; r < 8; ++r) { mrun[r] = -1e30f; lrun[r] = 0.0f; }

  _Float16* prow = &sP[wave * 16 * PSTRIDE];

  // ---- KV prefetch state: cooperative layout (4 threads x 16 floats per row) ----
  const int srow = tid >> 2;            // 0..63
  const int dcol = (tid & 3) << 4;      // 0,16,32,48
  const float* kvbase = qkv + baseBT + (size_t)(NH * HD) + h * HD + dcol;

  float4 kr[4], vr[4];
  {
    const float* kp = kvbase + (size_t)srow * TSTRIDE;
    const float* vp = kp + NH * HD;
#pragma unroll
    for (int i = 0; i < 4; ++i) {
      kr[i] = reinterpret_cast<const float4*>(kp)[i];
      vr[i] = reinterpret_cast<const float4*>(vp)[i];
    }
  }
  int mval = (tid < BN) ? kpm[b * SEQ + tid] : 0;

  for (int s0 = 0; s0 < SEQ; s0 += BN) {
    // ---- stage current tile (registers -> f16 LDS) ----
    {
      float kf[16] = {kr[0].x, kr[0].y, kr[0].z, kr[0].w,
                      kr[1].x, kr[1].y, kr[1].z, kr[1].w,
                      kr[2].x, kr[2].y, kr[2].z, kr[2].w,
                      kr[3].x, kr[3].y, kr[3].z, kr[3].w};
      H8 p0, p1;
#pragma unroll
      for (int i = 0; i < 8; ++i) { p0.h[i] = (_Float16)kf[i]; p1.h[i] = (_Float16)kf[i + 8]; }
      *reinterpret_cast<uint4*>(&sK[srow * KSTRIDE + dcol])     = p0.u;
      *reinterpret_cast<uint4*>(&sK[srow * KSTRIDE + dcol + 8]) = p1.u;

      float vf[16] = {vr[0].x, vr[0].y, vr[0].z, vr[0].w,
                      vr[1].x, vr[1].y, vr[1].z, vr[1].w,
                      vr[2].x, vr[2].y, vr[2].z, vr[2].w,
                      vr[3].x, vr[3].y, vr[3].z, vr[3].w};
#pragma unroll
      for (int i = 0; i < 16; ++i)
        sV[(dcol + i) * KSTRIDE + srow] = (_Float16)vf[i];

      if (tid < BN)
        sMask[tid] = mval ? 0.0f : MADD;
    }
    __syncthreads();

    // ---- prefetch next tile into registers (clamped; overlaps WMMAs below) ----
    {
      int sn = s0 + BN;
      if (sn >= SEQ) sn = 0;                       // harmless re-read on last iter
      const float* kp = kvbase + (size_t)(sn + srow) * TSTRIDE;
      const float* vp = kp + NH * HD;
#pragma unroll
      for (int i = 0; i < 4; ++i) {
        kr[i] = reinterpret_cast<const float4*>(kp)[i];
        vr[i] = reinterpret_cast<const float4*>(vp)[i];
      }
      mval = (tid < BN) ? kpm[b * SEQ + sn + tid] : 0;
    }

    // ---- scores: 16 x 64 strip = 4 tiles, K-dim 64 = 2 WMMA each ----
    float madd[4];
#pragma unroll
    for (int nc = 0; nc < 4; ++nc) madd[nc] = sMask[nc * 16 + l16];

    v8f Sacc[4];
#pragma unroll
    for (int nc = 0; nc < 4; ++nc) {
      const _Float16* kp = &sK[(nc * 16 + l16) * KSTRIDE + hl * 8];
      v16h b0 = ld16h2(kp, kp + 16);           // d = 0..31 chunk
      v16h b1 = ld16h2(kp + 32, kp + 48);      // d = 32..63 chunk
      v8f acc = (v8f){0, 0, 0, 0, 0, 0, 0, 0};
      acc = wmma_f16(qa[0], b0, acc);
      acc = wmma_f16(qa[1], b1, acc);
#pragma unroll
      for (int r = 0; r < 8; ++r) acc[r] += madd[nc];  // mask, log2 domain
      Sacc[nc] = acc;
    }

    // ---- online softmax (rows live in 16-lane halves) ----
    float mnew[8], alpha[8], psum[8];
#pragma unroll
    for (int r = 0; r < 8; ++r) {
      float mx = fmaxf(fmaxf(Sacc[0][r], Sacc[1][r]),
                       fmaxf(Sacc[2][r], Sacc[3][r]));
      mx = fmaxf(mx, __shfl_xor(mx, 1, 32));
      mx = fmaxf(mx, __shfl_xor(mx, 2, 32));
      mx = fmaxf(mx, __shfl_xor(mx, 4, 32));
      mx = fmaxf(mx, __shfl_xor(mx, 8, 32));
      mnew[r]  = fmaxf(mrun[r], mx);
      alpha[r] = exp2f(mrun[r] - mnew[r]);
      mrun[r]  = mnew[r];
      psum[r]  = 0.0f;
    }
#pragma unroll
    for (int dc = 0; dc < 4; ++dc)
#pragma unroll
      for (int r = 0; r < 8; ++r) Oacc[dc][r] *= alpha[r];

    // ---- P = exp2(S - m), f16 into wave-private LDS (C-layout -> row-major) ----
#pragma unroll
    for (int nc = 0; nc < 4; ++nc)
#pragma unroll
      for (int r = 0; r < 8; ++r) {
        float p = exp2f(Sacc[nc][r] - mnew[r]);
        psum[r] += p;
        prow[(r + hl * 8) * PSTRIDE + nc * 16 + l16] = (_Float16)p;
      }
#pragma unroll
    for (int r = 0; r < 8; ++r) {
      float s = psum[r];
      s += __shfl_xor(s, 1, 32);
      s += __shfl_xor(s, 2, 32);
      s += __shfl_xor(s, 4, 32);
      s += __shfl_xor(s, 8, 32);
      lrun[r] = lrun[r] * alpha[r] + s;
    }

    // ---- O += P x V : A-frags from sP, B-frags from transposed sV ----
#pragma unroll
    for (int kb = 0; kb < 2; ++kb) {
      const _Float16* pp = prow + l16 * PSTRIDE + kb * 32 + hl * 8;
      v16h pa = ld16h2(pp, pp + 16);           // A layout: chunks +0 and +16
#pragma unroll
      for (int dc = 0; dc < 4; ++dc) {
        const _Float16* vp = &sV[(dc * 16 + l16) * KSTRIDE + kb * 32 + hl * 8];
        v16h vb = ld16h2(vp, vp + 16);
        Oacc[dc] = wmma_f16(pa, vb, Oacc[dc]);
      }
    }
    __syncthreads();   // sK/sV reused next tile
  }

  // ---- epilogue: O / l, f32 store to (B,S,H,D) ----
#pragma unroll
  for (int r = 0; r < 8; ++r) {
    const float inv = 1.0f / lrun[r];
    const int   t   = tbase + r + hl * 8;
    const size_t ob = (((size_t)b * SEQ + t) * NH + h) * HD;
#pragma unroll
    for (int dc = 0; dc < 4; ++dc)
      out[ob + dc * 16 + l16] = Oacc[dc][r] * inv;
  }
}

extern "C" void kernel_launch(void* const* d_in, const int* in_sizes, int n_in,
                              void* d_out, int out_size, void* d_ws, size_t ws_size,
                              hipStream_t stream) {
  (void)in_sizes; (void)n_in; (void)out_size; (void)d_ws; (void)ws_size;
  const float* qkv = (const float*)d_in[0];
  const int*   kpm = (const int*)d_in[1];
  float*       out = (float*)d_out;
  dim3 grid(SEQ / BM, NH, NB);   // 16 x 16 x 4 workgroups
  fa_fwd_kernel<<<grid, dim3(256), 0, stream>>>(qkv, kpm, out);
}
